// ParallelEGAT_60284160967031
// MI455X (gfx1250) — matrix operand order repaired and checked by reading, not compiled, not run on hardware
//
#include <hip/hip_runtime.h>
#include <hip/hip_bf16.h>
#include <hip/hip_fp16.h>

typedef __attribute__((ext_vector_type(16))) _Float16 v16h;
typedef __attribute__((ext_vector_type(8)))  float    v8f;

#define N_NODES   50000
#define N_EDGES   800000
#define DIMS      5
#define IN_C      32
#define OUT_C     32
#define FEAT      (DIMS * IN_C)   // 160
#define ATT_W     (2 * OUT_C + 1) // 65
#define NEG_SLOPE 0.2f

// ---------------------------------------------------------------------------
// float <-> ordered-uint key (monotone), for atomicMax-based segment max
// ---------------------------------------------------------------------------
__device__ __forceinline__ unsigned f2key(float f) {
  unsigned u = __float_as_uint(f);
  return (u & 0x80000000u) ? ~u : (u | 0x80000000u);
}
__device__ __forceinline__ float key2f(unsigned k) {
  unsigned u = (k & 0x80000000u) ? (k & 0x7FFFFFFFu) : ~k;
  return __uint_as_float(u);
}

// ---------------------------------------------------------------------------
// Kernel 1: h[n,d,o] = sum_i x[n,i,d] * W[d,i,o]  via v_wmma_f32_16x16x32_f16
// One wave computes a 16-node tile; per dim d: one WMMA per 16-wide col half.
// A (16x32 f16): lane(half,m); pairs per VGPR per ISA layout.
// B (32x16 f16): lane(half,n) holds K=8*half..8*half+7, 16+8*half..16+8*half+7 across 8 VGPRs.
// C (16x16 f32): lane(half,n), VGPR v -> row M = v + 8*half.
// ---------------------------------------------------------------------------
__global__ __launch_bounds__(256) void egat_gemm_h(
    const float* __restrict__ x, const float* __restrict__ W,
    float* __restrict__ h)
{
  const int lane = threadIdx.x & 31;
  const int wave = threadIdx.x >> 5;
  const int tile = blockIdx.x * 8 + wave;   // 16 nodes per tile
  const int n0   = tile * 16;
  if (n0 >= N_NODES) return;                // wave-uniform: EXEC stays all-1s
  const int half = lane >> 4;
  const int mn   = lane & 15;

  for (int d = 0; d < DIMS; ++d) {
    // ---- load A tile (rows = nodes, K = input channel) as f16 ----
    v16h a;
#pragma unroll
    for (int v = 0; v < 8; ++v) {
      const int k0 = 8 * half + 2 * (v & 3) + 16 * (v >> 2);
      const float a0 = x[(size_t)(n0 + mn) * FEAT + (size_t)k0 * DIMS + d];
      const float a1 = x[(size_t)(n0 + mn) * FEAT + (size_t)(k0 + 1) * DIMS + d];
      a[2 * v]     = (_Float16)a0;
      a[2 * v + 1] = (_Float16)a1;
    }
#pragma unroll
    for (int ch = 0; ch < 2; ++ch) {
      // ---- load B tile (K x 16 cols of W_d) as f16 ----
      v16h b;
#pragma unroll
      for (int v = 0; v < 8; ++v) {
        const int k = 8 * half + 2 * (v & 3) + 16 * (v >> 2);
        b[2 * v]     = (_Float16)W[(size_t)d * IN_C * OUT_C + (size_t)k * OUT_C + ch * 16 + mn];
        b[2 * v + 1] = (_Float16)W[(size_t)d * IN_C * OUT_C + (size_t)(k + 1) * OUT_C + ch * 16 + mn];
      }
      v8f c = {0.f, 0.f, 0.f, 0.f, 0.f, 0.f, 0.f, 0.f};
      // D = A(16x32) * B(32x16) + C ; full K in one instruction
      c = __builtin_amdgcn_wmma_f32_16x16x32_f16(
          /*neg_a=*/false, a, /*neg_b=*/false, b,
          /*c_mod=*/(short)0, c, /*reuse_a=*/false, /*reuse_b=*/false);
#pragma unroll
      for (int v = 0; v < 8; ++v) {
        h[(size_t)(n0 + v + 8 * half) * FEAT + (size_t)d * OUT_C + ch * 16 + mn] = c[v];
      }
    }
  }
}

// ---------------------------------------------------------------------------
// Kernel 2: zero a float/u32 buffer (out accumulator, smax keys, ssum)
// ---------------------------------------------------------------------------
__global__ void egat_zero(float* __restrict__ p, int n) {
  int i = blockIdx.x * blockDim.x + threadIdx.x;
  if (i < n) p[i] = 0.0f;
}

// ---------------------------------------------------------------------------
// Kernel 3: per-(edge,dim) attention score + leaky relu + segment max
// ---------------------------------------------------------------------------
__global__ void egat_score(const float* __restrict__ h,
                           const long long* __restrict__ ei,
                           const float* __restrict__ edge_attr,
                           const float* __restrict__ att,
                           float* __restrict__ score,
                           unsigned* __restrict__ smax)
{
  const int t = blockIdx.x * blockDim.x + threadIdx.x;
  if (t >= N_EDGES * DIMS) return;
  const int e = t / DIMS;
  const int d = t - e * DIMS;
  const long long row = ei[e];
  const long long col = ei[N_EDGES + e];
  const float* hi = h + (size_t)row * FEAT + (size_t)d * OUT_C;
  const float* hj = h + (size_t)col * FEAT + (size_t)d * OUT_C;
  const float* ai = att + (size_t)d * ATT_W;
  const float* aj = ai + OUT_C;
  float s = edge_attr[(size_t)e * DIMS + d] * ai[2 * OUT_C];
#pragma unroll
  for (int o = 0; o < OUT_C; ++o)
    s += hi[o] * ai[o] + hj[o] * aj[o];
  s = (s > 0.f) ? s : NEG_SLOPE * s;
  score[t] = s;
  atomicMax(smax + (size_t)row * DIMS + d, f2key(s));
}

// ---------------------------------------------------------------------------
// Kernel 4: escore = exp(score - smax[row]); segment sum (in-place escore)
// ---------------------------------------------------------------------------
__global__ void egat_exp(const long long* __restrict__ ei,
                         const unsigned* __restrict__ smax,
                         float* __restrict__ score,   // in: score, out: escore
                         float* __restrict__ ssum)
{
  const int t = blockIdx.x * blockDim.x + threadIdx.x;
  if (t >= N_EDGES * DIMS) return;
  const int e = t / DIMS;
  const int d = t - e * DIMS;
  const long long row = ei[e];
  const float m  = key2f(smax[(size_t)row * DIMS + d]);
  const float es = __expf(score[t] - m);
  score[t] = es;
  atomicAdd(ssum + (size_t)row * DIMS + d, es);
}

// ---------------------------------------------------------------------------
// Kernel 5: alpha = escore / (ssum[row] + eps)  -> written straight to d_out
// ---------------------------------------------------------------------------
__global__ void egat_alpha(const long long* __restrict__ ei,
                           const float* __restrict__ escore,
                           const float* __restrict__ ssum,
                           float* __restrict__ alpha)
{
  const int t = blockIdx.x * blockDim.x + threadIdx.x;
  if (t >= N_EDGES * DIMS) return;
  const int e = t / DIMS;
  const int d = t - e * DIMS;
  const long long row = ei[e];
  alpha[t] = escore[t] / (ssum[(size_t)row * DIMS + d] + 1e-16f);
}

// ---------------------------------------------------------------------------
// Kernel 6: out[row,d,:] += alpha[e,d] * h[col,d,:]  (per-(edge,channel) thread)
// ---------------------------------------------------------------------------
__global__ void egat_scatter(const long long* __restrict__ ei,
                             const float* __restrict__ alpha,
                             const float* __restrict__ h,
                             float* __restrict__ out)
{
  const long long t = (long long)blockIdx.x * blockDim.x + threadIdx.x;
  if (t >= (long long)N_EDGES * 32) return;
  const int e = (int)(t >> 5);
  const int o = (int)(t & 31);
  const long long row = ei[e];
  const long long col = ei[N_EDGES + e];
#pragma unroll
  for (int d = 0; d < DIMS; ++d) {
    const float v = alpha[(size_t)e * DIMS + d] *
                    h[(size_t)col * FEAT + (size_t)d * OUT_C + o];
    atomicAdd(out + (size_t)row * FEAT + (size_t)d * OUT_C + o, v);
  }
}

// ---------------------------------------------------------------------------
extern "C" void kernel_launch(void* const* d_in, const int* in_sizes, int n_in,
                              void* d_out, int out_size, void* d_ws, size_t ws_size,
                              hipStream_t stream)
{
  const float*     x         = (const float*)d_in[0];
  const long long* ei        = (const long long*)d_in[1];
  const float*     edge_attr = (const float*)d_in[2];
  const float*     W         = (const float*)d_in[3];
  const float*     att       = (const float*)d_in[4];

  // d_out layout: [ out : N*160 f32 ][ alpha : E*5 f32 ][ edge_index : 2E i64 ]
  float* out   = (float*)d_out;
  float* alpha = out + (size_t)N_NODES * FEAT;
  char*  eiOut = (char*)(alpha + (size_t)N_EDGES * DIMS);

  // d_ws layout: [ h : N*160 f32 ][ score/escore : E*5 f32 ][ smax : N*5 u32 ][ ssum : N*5 f32 ]
  float*    h     = (float*)d_ws;
  float*    score = h + (size_t)N_NODES * FEAT;
  unsigned* smax  = (unsigned*)(score + (size_t)N_EDGES * DIMS);
  float*    ssum  = (float*)(smax + (size_t)N_NODES * DIMS);

  // 1) h = x @ W  (WMMA; 8 waves/block, 16-node tile per wave)
  {
    const int tiles  = (N_NODES + 15) / 16;   // 3125 (exact)
    const int blocks = (tiles + 7) / 8;       // 391
    egat_gemm_h<<<blocks, 256, 0, stream>>>(x, W, h);
  }

  // 2) zero accumulators (must run every call: graph replay re-accumulates)
  egat_zero<<<(N_NODES * FEAT + 255) / 256, 256, 0, stream>>>(out, N_NODES * FEAT);
  egat_zero<<<(N_NODES * DIMS + 255) / 256, 256, 0, stream>>>((float*)smax, N_NODES * DIMS);
  egat_zero<<<(N_NODES * DIMS + 255) / 256, 256, 0, stream>>>(ssum, N_NODES * DIMS);

  const int ED = N_EDGES * DIMS;
  // 3) scores + segment max
  egat_score<<<(ED + 255) / 256, 256, 0, stream>>>(h, ei, edge_attr, att, score, smax);
  // 4) exp + segment sum
  egat_exp<<<(ED + 255) / 256, 256, 0, stream>>>(ei, smax, score, ssum);
  // 5) alpha -> d_out
  egat_alpha<<<(ED + 255) / 256, 256, 0, stream>>>(ei, score, ssum, alpha);
  // 6) scatter-accumulate out
  {
    const long long tot = (long long)N_EDGES * 32;
    egat_scatter<<<(int)((tot + 255) / 256), 256, 0, stream>>>(ei, alpha, h, out);
  }

  // 7) edge_index passthrough (clamped to reported out_size, d2d async is
  //    graph-capture safe)
  {
    const size_t usedBytes = ((size_t)N_NODES * FEAT + (size_t)N_EDGES * DIMS) * sizeof(float);
    const size_t wantBytes = (size_t)2 * N_EDGES * sizeof(long long);
    const size_t outBytes  = (size_t)out_size * sizeof(float);
    if (outBytes > usedBytes) {
      const size_t cpyBytes = (outBytes - usedBytes) < wantBytes ? (outBytes - usedBytes)
                                                                 : wantBytes;
      hipMemcpyAsync(eiOut, d_in[1], cpyBytes, hipMemcpyDeviceToDevice, stream);
    }
  }
  (void)in_sizes; (void)n_in; (void)ws_size;
}
